// Basic_EGNN_81836306858707
// MI455X (gfx1250) — matrix operand order repaired and checked by reading, not compiled, not run on hardware
//
#include <hip/hip_runtime.h>
#include <hip/hip_bf16.h>
#include <math.h>

// ---------------------------------------------------------------------------
// Problem constants (from reference): R=2, BS=128, NATM=81, NORB=240
// Assembled FC1 input vector: 4*59*59 (convs) + 81*2 (egnn) + 1 (egfn1) = 14087
// Padded for WMMA f16 (K multiple of 32): 14112. N padded 60 -> 64.
// ---------------------------------------------------------------------------
#define RR     2
#define BS     128
#define NATM   81
#define NORB   240
#define KIN    14087
#define KPAD   14112
#define NPAD   64
#define KCHUNK 96                    // halfs per staged weight chunk (3 k-steps)
#define NKC    (KPAD / KCHUNK)       // 147 chunks

typedef _Float16 v16h __attribute__((ext_vector_type(16)));
typedef _Float16 v8h  __attribute__((ext_vector_type(8)));
typedef float    v8f  __attribute__((ext_vector_type(8)));

__device__ __forceinline__ float silu_f(float x) {
    return x / (1.0f + __expf(-x));
}
__device__ __forceinline__ float lrelu_f(float x) {
    return x > 0.0f ? x : 0.01f * x;
}

// ---------------------------------------------------------------------------
// CDNA5 async global->LDS copy (ASYNCcnt path), guarded for toolchain support.
// Builtin signature (from hipcc diagnostic): param0 = AS(1) ptr to gcc-style
// int __vector(4) (global source), param1 = AS(3) ptr (LDS dest), imm offset,
// imm cpol.
// ---------------------------------------------------------------------------
#if __has_builtin(__builtin_amdgcn_global_load_async_to_lds_b128)
#define HAVE_ASYNC_LDS 1
typedef int gv4i __attribute__((__vector_size__(4 * sizeof(int))));
typedef __attribute__((address_space(1))) gv4i* gv4i_g;
typedef __attribute__((address_space(3))) gv4i* gv4i_l;
#endif

__device__ __forceinline__ void copy16_g2l(const _Float16* g, _Float16* l) {
#ifdef HAVE_ASYNC_LDS
    __builtin_amdgcn_global_load_async_to_lds_b128(
        (gv4i_g)(uintptr_t)g,
        (gv4i_l)(unsigned)(uintptr_t)l,
        0, 0);
#else
    *(v8h*)l = *(const v8h*)g;
#endif
}

__device__ __forceinline__ void wait_async0() {
#if __has_builtin(__builtin_amdgcn_s_wait_asynccnt)
    __builtin_amdgcn_s_wait_asynccnt(0);
#elif defined(HAVE_ASYNC_LDS)
    asm volatile("s_wait_asynccnt 0" ::: "memory");
#endif
}

// ---------------------------------------------------------------------------
// CDNA5 Tensor Data Mover: 1-D tile load, global -> LDS (TENSORcnt path).
// D# packing per CDNA5 ISA. Guarded; arity per
// toolchain (ROCm7.2/clang-22: 5 args, amdgpu-toolchain/clang-23: 6 args).
// ---------------------------------------------------------------------------
#if __has_builtin(__builtin_amdgcn_tensor_load_to_lds) && \
    __has_builtin(__builtin_amdgcn_s_wait_tensorcnt)
#define HAVE_TDM 1
typedef unsigned int u32x4 __attribute__((ext_vector_type(4)));
typedef int          i32x4 __attribute__((ext_vector_type(4)));
typedef int          i32x8 __attribute__((ext_vector_type(8)));

__device__ __forceinline__ void tdm_load_1d_f32(const float* gsrc, void* ldst,
                                                int ne) {
    unsigned long long ga = (unsigned long long)(uintptr_t)gsrc;
    unsigned lds = (unsigned)(uintptr_t)ldst;   // low 32 bits = LDS byte offset
    u32x4 g0;
    g0.x = 1u;                                        // count=1 (user mode)
    g0.y = lds;                                       // lds_addr
    g0.z = (unsigned)(ga & 0xFFFFFFFFull);            // global_addr[31:0]
    g0.w = (unsigned)((ga >> 32) & 0x01FFFFFFull)     // global_addr[56:32]
         | (2u << 30);                                // type=2 ("image")
    i32x8 g1;
    g1[0] = (int)(2u << 16);                          // data_size=4B, mask=0
    g1[1] = (int)(((unsigned)ne & 0xFFFFu) << 16);    // tensor_dim0[15:0]
    g1[2] = (int)(((unsigned)ne >> 16) & 0xFFFFu);    // tensor_dim0[31:16], dim1=0
    g1[3] = (int)(((unsigned)ne & 0xFFFFu) << 16);    // tile_dim0 = ne
    g1[4] = 0;                                        // tile_dim1/2 = 0 (unused)
    g1[5] = ne;                                       // tensor_dim0_stride[31:0]
    g1[6] = 0;
    g1[7] = 0;
    i32x4 z4 = (i32x4)0;
#if __clang_major__ >= 23
    i32x8 z8 = (i32x8)0;
    __builtin_amdgcn_tensor_load_to_lds(g0, g1, z4, z4, z8, 0);
#else
    __builtin_amdgcn_tensor_load_to_lds(g0, g1, z4, z4, 0);
#endif
}
#endif

// ---------------------------------------------------------------------------
// Kernel 1: convert fc1_w (60 x 14087, f32) -> padded f16 weight (64 x 14112)
// ---------------------------------------------------------------------------
__global__ void convert_w_kernel(const float* __restrict__ fc1_w,
                                 _Float16* __restrict__ Wh) {
    int idx = blockIdx.x * blockDim.x + threadIdx.x;
    if (idx >= NPAD * KPAD) return;
    int n = idx / KPAD;
    int k = idx - n * KPAD;
    float v = (n < 60 && k < KIN) ? fc1_w[n * KIN + k] : 0.0f;
    Wh[idx] = (_Float16)v;
}

// ---------------------------------------------------------------------------
// Kernel 2: fused conv block.  One block per (r, b, src in {ovlp,h0}).
// conv1(2x2)+bias+leaky+maxpool2 -> s1[2][119][119] (f16 in LDS),
// conv2(2x2)+bias+leaky+maxpool2 -> X[..][src*6962 + oc*3481 + y*59 + x] (f16)
// ---------------------------------------------------------------------------
__global__ __launch_bounds__(512)
void conv_kernel(const float* __restrict__ ovlp, const float* __restrict__ h0,
                 const float* __restrict__ c1w, const float* __restrict__ c1b,
                 const float* __restrict__ c2w, const float* __restrict__ c2b,
                 _Float16* __restrict__ X) {
    const int gid = blockIdx.x;            // ((r*BS + b)*2 + s)
    const int s   = gid & 1;
    const int rb  = gid >> 1;
    const float* img = (s == 0 ? ovlp : h0) + (size_t)rb * NORB * NORB;

    __shared__ _Float16 s1[2 * 119 * 119];   // 56,644 bytes

    float w1[2][2][2], b1[2];
#pragma unroll
    for (int c = 0; c < 2; ++c) {
        b1[c] = c1b[c];
#pragma unroll
        for (int dy = 0; dy < 2; ++dy)
#pragma unroll
            for (int dx = 0; dx < 2; ++dx)
                w1[c][dy][dx] = c1w[(c * 2 + dy) * 2 + dx];
    }

    // -------- stage 1: conv1 + leaky + 2x2 maxpool --------
    for (int idx = threadIdx.x; idx < 2 * 119 * 119; idx += blockDim.x) {
        int c   = idx / (119 * 119);
        int rem = idx - c * (119 * 119);
        int y   = rem / 119;
        int x   = rem - y * 119;
        int py0 = 2 * y, px0 = 2 * x;
        __builtin_prefetch(img + (py0 + 2) * NORB + px0, 0, 1);
        float mx = -3.0e38f;
#pragma unroll
        for (int oy = 0; oy < 2; ++oy) {
#pragma unroll
            for (int ox = 0; ox < 2; ++ox) {
                const float* p = img + (py0 + oy) * NORB + (px0 + ox);
                float v = b1[c]
                        + p[0]        * w1[c][0][0] + p[1]        * w1[c][0][1]
                        + p[NORB]     * w1[c][1][0] + p[NORB + 1] * w1[c][1][1];
                v = lrelu_f(v);
                mx = fmaxf(mx, v);
            }
        }
        s1[idx] = (_Float16)mx;
    }
    __syncthreads();

    float w2[2][2][2][2], b2[2];
#pragma unroll
    for (int oc = 0; oc < 2; ++oc) {
        b2[oc] = c2b[oc];
#pragma unroll
        for (int ic = 0; ic < 2; ++ic)
#pragma unroll
            for (int dy = 0; dy < 2; ++dy)
#pragma unroll
                for (int dx = 0; dx < 2; ++dx)
                    w2[oc][ic][dy][dx] = c2w[(((oc * 2 + ic) * 2 + dy) * 2 + dx)];
    }

    // -------- stage 2: conv2 + leaky + 2x2 maxpool -> f16 into GEMM input ----
    _Float16* xrow = X + (size_t)rb * KPAD + s * 6962;
    for (int idx = threadIdx.x; idx < 2 * 59 * 59; idx += blockDim.x) {
        int oc  = idx / 3481;
        int rem = idx - oc * 3481;
        int y   = rem / 59;
        int x   = rem - y * 59;
        int py0 = 2 * y, px0 = 2 * x;
        float mx = -3.0e38f;
#pragma unroll
        for (int oy = 0; oy < 2; ++oy) {
#pragma unroll
            for (int ox = 0; ox < 2; ++ox) {
                float v = b2[oc];
#pragma unroll
                for (int ic = 0; ic < 2; ++ic) {
                    const _Float16* q = s1 + ic * (119 * 119)
                                      + (py0 + oy) * 119 + (px0 + ox);
                    v += (float)q[0]   * w2[oc][ic][0][0]
                       + (float)q[1]   * w2[oc][ic][0][1]
                       + (float)q[119] * w2[oc][ic][1][0]
                       + (float)q[120] * w2[oc][ic][1][1];
                }
                v = lrelu_f(v);
                mx = fmaxf(mx, v);
            }
        }
        xrow[idx] = (_Float16)mx;   // idx == oc*3481 + y*59 + x
    }
}

// ---------------------------------------------------------------------------
// Kernel 3: EGNN node features. One block per (r,b). Coordinate-update branch
// (Wc1/Wc2) of the reference is dead code w.r.t. the output -> skipped.
// Per-block input slabs are DMA'd global->LDS by the Tensor Data Mover when
// available (TENSORcnt), else plain loads.
// ---------------------------------------------------------------------------
__global__ __launch_bounds__(256)
void egnn_kernel(const float* __restrict__ xyz,  const float* __restrict__ cn,
                 const float* __restrict__ edisp,const float* __restrict__ egfn1,
                 const float* __restrict__ We1,  const float* __restrict__ be1,
                 const float* __restrict__ We2,  const float* __restrict__ be2,
                 const float* __restrict__ Wn1,  const float* __restrict__ bn1,
                 const float* __restrict__ Wn2,  const float* __restrict__ bn2,
                 _Float16* __restrict__ X) {
    const int rb  = blockIdx.x;          // r*BS + b
    const int tid = threadIdx.x;

    __shared__ __attribute__((aligned(16))) float sC[NATM * 3];  // xyz, [a*3+c]
    __shared__ __attribute__((aligned(16))) float sF0[NATM];     // cn
    __shared__ __attribute__((aligned(16))) float sF1[NATM];     // edisp
    __shared__ float mpart[NATM][3][16];
    __shared__ float sWe1[50], sbe1[10], sWe2[160], sbe2[16];
    __shared__ float sWn1[72], sbn1[4],  sWn2[8],   sbn2[2];

    for (int t = tid; t < 50;  t += 256) sWe1[t] = We1[t];
    for (int t = tid; t < 10;  t += 256) sbe1[t] = be1[t];
    for (int t = tid; t < 160; t += 256) sWe2[t] = We2[t];
    for (int t = tid; t < 16;  t += 256) sbe2[t] = be2[t];
    for (int t = tid; t < 72;  t += 256) sWn1[t] = Wn1[t];
    for (int t = tid; t < 4;   t += 256) sbn1[t] = bn1[t];
    for (int t = tid; t < 8;   t += 256) sWn2[t] = Wn2[t];
    for (int t = tid; t < 2;   t += 256) sbn2[t] = bn2[t];

#ifdef HAVE_TDM
    if (tid < 32) {   // wave 0 issues three 1-D TDM descriptors
        tdm_load_1d_f32(xyz   + (size_t)rb * NATM * 3, sC,  NATM * 3);
        tdm_load_1d_f32(cn    + (size_t)rb * NATM,     sF0, NATM);
        tdm_load_1d_f32(edisp + (size_t)rb * NATM,     sF1, NATM);
        __builtin_amdgcn_s_wait_tensorcnt(0);
    }
#else
    for (int t = tid; t < NATM * 3; t += 256)
        sC[t] = xyz[(size_t)rb * NATM * 3 + t];
    for (int t = tid; t < NATM; t += 256) {
        sF0[t] = cn[(size_t)rb * NATM + t];
        sF1[t] = edisp[(size_t)rb * NATM + t];
    }
#endif
    // zero the tail padding [KIN, KPAD) of this row while we're here
    _Float16* xrow = X + (size_t)rb * KPAD;
    if (tid < KPAD - KIN) xrow[KIN + tid] = (_Float16)0.0f;
    __syncthreads();

    // 243 workers: 3 edge-slots per atom i, no atomics.
    if (tid < 3 * NATM) {
        const int i = tid / 3, slot = tid - 3 * i;
        float acc[16];
#pragma unroll
        for (int k = 0; k < 16; ++k) acc[k] = 0.0f;
        const float ci0 = sC[i * 3 + 0], ci1 = sC[i * 3 + 1], ci2 = sC[i * 3 + 2];
        const float fi0 = sF0[i], fi1 = sF1[i];
        for (int j = slot; j < NATM; j += 3) {
            float dx = ci0 - sC[j * 3 + 0];
            float dy = ci1 - sC[j * 3 + 1];
            float dz = ci2 - sC[j * 3 + 2];
            float in5[5] = { fi0, fi1, sF0[j], sF1[j],
                             dx * dx + dy * dy + dz * dz };
            float h10[10];
#pragma unroll
            for (int k = 0; k < 10; ++k) {
                float sacc = sbe1[k];
#pragma unroll
                for (int q = 0; q < 5; ++q) sacc += sWe1[k * 5 + q] * in5[q];
                h10[k] = silu_f(sacc);
            }
#pragma unroll
            for (int k = 0; k < 16; ++k) {
                float sacc = sbe2[k];
#pragma unroll
                for (int q = 0; q < 10; ++q) sacc += sWe2[k * 10 + q] * h10[q];
                acc[k] += silu_f(sacc);
            }
        }
#pragma unroll
        for (int k = 0; k < 16; ++k) mpart[i][slot][k] = acc[k];
    }
    __syncthreads();

    // node MLP per atom
    if (tid < NATM) {
        const int i = tid;
        float in18[18];
        in18[0] = sF0[i];
        in18[1] = sF1[i];
#pragma unroll
        for (int k = 0; k < 16; ++k)
            in18[2 + k] = mpart[i][0][k] + mpart[i][1][k] + mpart[i][2][k];
        float h4[4];
#pragma unroll
        for (int k = 0; k < 4; ++k) {
            float sacc = sbn1[k];
#pragma unroll
            for (int q = 0; q < 18; ++q) sacc += sWn1[k * 18 + q] * in18[q];
            h4[k] = silu_f(sacc);
        }
        float o0 = sbn2[0] + sF0[i];
        float o1 = sbn2[1] + sF1[i];
#pragma unroll
        for (int q = 0; q < 4; ++q) {
            o0 += sWn2[q]     * h4[q];
            o1 += sWn2[4 + q] * h4[q];
        }
        xrow[13924 + 2 * i]     = (_Float16)o0;
        xrow[13924 + 2 * i + 1] = (_Float16)o1;
    }
    if (tid == 255) xrow[14086] = (_Float16)egfn1[rb];
}

// ---------------------------------------------------------------------------
// Kernel 4: zero the 128-element output accumulator.
// ---------------------------------------------------------------------------
__global__ void zero_out_kernel(float* __restrict__ out) {
    if (threadIdx.x < BS) out[threadIdx.x] = 0.0f;
}

// ---------------------------------------------------------------------------
// Kernel 5: FC head with WMMA.  One block per r; 8 waves = 8 M-tiles of the
// M=128, K=14112, N=64 GEMM (v_wmma_f32_16x16x32_f16).  All waves share the
// same B panel, so it is staged once per block in LDS via double-buffered
// async global->LDS copies (ASYNCcnt) and consumed with ds_load_b128.
// Epilogue: +fc1_b, leaky, FC2(30x60), leaky, FC3(1x30), *nu, atomic add.
// ---------------------------------------------------------------------------
__global__ __launch_bounds__(256)
void fc_kernel(const _Float16* __restrict__ X, const _Float16* __restrict__ Wh,
               const float* __restrict__ fc1_b,
               const float* __restrict__ fc2_w, const float* __restrict__ fc2_b,
               const float* __restrict__ fc3_w, const float* __restrict__ fc3_b,
               const float* __restrict__ nu, float* __restrict__ out) {
    __shared__ __attribute__((aligned(16))) _Float16 Bs[2][NPAD * KCHUNK]; // 24KB
    __shared__ float y1[BS][NPAD];                                         // 32KB

    const int r    = blockIdx.x;
    const int wave = threadIdx.x >> 5;       // M-tile index, 0..7
    const int lane = threadIdx.x & 31;

    const _Float16* Xr = X + (size_t)r * BS * KPAD;

    // A fragment 16x32 f16: lane<16 -> row m0+lane,  K {0..7, 16..23}
    //                       lane>=16-> row m0+lane-16, K {8..15, 24..31}
    const int m      = wave * 16 + (lane & 15);
    const int aKbase = (lane < 16) ? 0 : 8;
    // B fragment 32x16 f16: lanes 0-15 -> K {0..7, 16..23} at N=lane; 16-31 -> K {8..15, 24..31}
    const int nidx   = lane & 15;
    const int bKbase = (lane < 16) ? 0 : 8;

    const _Float16* arow = Xr + (size_t)m * KPAD + aKbase;

    v8f acc[4];
#pragma unroll
    for (int nt = 0; nt < 4; ++nt) acc[nt] = (v8f)(0.0f);

    // stage chunk 0 (uniform trip count: 768 packets / 256 threads = 3 each)
    for (int p = threadIdx.x; p < NPAD * (KCHUNK / 8); p += 256) {
        int n = p / (KCHUNK / 8), c = p - n * (KCHUNK / 8);
        copy16_g2l(Wh + (size_t)n * KPAD + c * 8, &Bs[0][n * KCHUNK + c * 8]);
    }
    wait_async0();
    __syncthreads();

    for (int kc = 0; kc < NKC; ++kc) {
        const int cur = kc & 1;
        if (kc + 1 < NKC) {   // async-prefetch next weight chunk into other buf
            for (int p = threadIdx.x; p < NPAD * (KCHUNK / 8); p += 256) {
                int n = p / (KCHUNK / 8), c = p - n * (KCHUNK / 8);
                copy16_g2l(Wh + (size_t)n * KPAD + (kc + 1) * KCHUNK + c * 8,
                           &Bs[cur ^ 1][n * KCHUNK + c * 8]);
            }
        }
#pragma unroll
        for (int ks = 0; ks < KCHUNK / 32; ++ks) {
            const int k0 = kc * KCHUNK + ks * 32;
            v8h alo = *(const v8h*)(arow + k0);
            v8h ahi = *(const v8h*)(arow + k0 + 16);
            v16h a;
#pragma unroll
            for (int t = 0; t < 8; ++t) { a[t] = alo[t]; a[t + 8] = ahi[t]; }
#pragma unroll
            for (int nt = 0; nt < 4; ++nt) {
                const _Float16* bp =
                    &Bs[cur][(nt * 16 + nidx) * KCHUNK + ks * 32 + bKbase];
                v8h blo = *(const v8h*)bp;
                v8h bhi = *(const v8h*)(bp + 16);
                v16h bf;
#pragma unroll
                for (int t = 0; t < 8; ++t) { bf[t] = blo[t]; bf[t + 8] = bhi[t]; }
                acc[nt] = __builtin_amdgcn_wmma_f32_16x16x32_f16(
                    /*neg_a=*/false, a, /*neg_b=*/false, bf,
                    /*c_mod=*/(short)0, acc[nt],
                    /*reuse_a=*/false, /*reuse_b=*/false);
            }
        }
        wait_async0();       // next chunk's DMA complete before buffer swap
        __syncthreads();     // all waves done reading Bs[cur]
    }

    // D layout: c[v] is element (m0 + v + 8*(lane>=16), n0 + (lane&15))
    const int mbase = wave * 16 + ((lane >= 16) ? 8 : 0);
#pragma unroll
    for (int nt = 0; nt < 4; ++nt)
#pragma unroll
        for (int v = 0; v < 8; ++v)
            y1[mbase + v][nt * 16 + nidx] = acc[nt][v];
    __syncthreads();

    // epilogue: one thread per batch row
    if (threadIdx.x < BS) {
        const int b = threadIdx.x;
        float v[60];
#pragma unroll
        for (int n = 0; n < 60; ++n) v[n] = lrelu_f(y1[b][n] + fc1_b[n]);
        float z3 = fc3_b[0];
        for (int j = 0; j < 30; ++j) {
            float sacc = fc2_b[j];
#pragma unroll
            for (int n = 0; n < 60; ++n) sacc += fc2_w[j * 60 + n] * v[n];
            z3 += fc3_w[j] * lrelu_f(sacc);
        }
        atomicAdd(&out[b], z3 * nu[b * RR + r]);   // nu is (BS, R)
    }
}

// ---------------------------------------------------------------------------
// Host-side launch.  Input order follows setup_inputs():
//  0 xyz, 1 cn, 2 edisp, 3 ovlp, 4 h0, 5 egfn1, 6 nu, 7 We1, 8 be1, 9 We2,
// 10 be2, 11 Wc1, 12 bc1, 13 Wc2, 14 bc2, 15 coors_scale, 16 Wn1, 17 bn1,
// 18 Wn2, 19 bn2, 20 conv1_w, 21 conv1_b, 22 conv2_w, 23 conv2_b,
// 24 fc1_w, 25 fc1_b, 26 fc2_w, 27 fc2_b, 28 fc3_w, 29 fc3_b
// ---------------------------------------------------------------------------
extern "C" void kernel_launch(void* const* d_in, const int* in_sizes, int n_in,
                              void* d_out, int out_size, void* d_ws, size_t ws_size,
                              hipStream_t stream) {
    const float* xyz     = (const float*)d_in[0];
    const float* cn      = (const float*)d_in[1];
    const float* edisp   = (const float*)d_in[2];
    const float* ovlp    = (const float*)d_in[3];
    const float* h0      = (const float*)d_in[4];
    const float* egfn1   = (const float*)d_in[5];
    const float* nu      = (const float*)d_in[6];
    const float* We1     = (const float*)d_in[7];
    const float* be1     = (const float*)d_in[8];
    const float* We2     = (const float*)d_in[9];
    const float* be2     = (const float*)d_in[10];
    const float* Wn1     = (const float*)d_in[16];
    const float* bn1     = (const float*)d_in[17];
    const float* Wn2     = (const float*)d_in[18];
    const float* bn2     = (const float*)d_in[19];
    const float* conv1_w = (const float*)d_in[20];
    const float* conv1_b = (const float*)d_in[21];
    const float* conv2_w = (const float*)d_in[22];
    const float* conv2_b = (const float*)d_in[23];
    const float* fc1_w   = (const float*)d_in[24];
    const float* fc1_b   = (const float*)d_in[25];
    const float* fc2_w   = (const float*)d_in[26];
    const float* fc2_b   = (const float*)d_in[27];
    const float* fc3_w   = (const float*)d_in[28];
    const float* fc3_b   = (const float*)d_in[29];
    float* out = (float*)d_out;

    // workspace layout: [ Xf16: R*BS*KPAD halfs | Wh: NPAD*KPAD halfs ]
    _Float16* Xf = (_Float16*)d_ws;
    _Float16* Wh = Xf + (size_t)RR * BS * KPAD;

    convert_w_kernel<<<(NPAD * KPAD + 255) / 256, 256, 0, stream>>>(fc1_w, Wh);
    conv_kernel<<<RR * BS * 2, 512, 0, stream>>>(ovlp, h0, conv1_w, conv1_b,
                                                 conv2_w, conv2_b, Xf);
    egnn_kernel<<<RR * BS, 256, 0, stream>>>(xyz, cn, edisp, egfn1,
                                             We1, be1, We2, be2,
                                             Wn1, bn1, Wn2, bn2, Xf);
    zero_out_kernel<<<1, 128, 0, stream>>>(out);
    fc_kernel<<<RR, 256, 0, stream>>>(Xf, Wh, fc1_b, fc2_w, fc2_b,
                                      fc3_w, fc3_b, nu, out);
}
